// MultiHeadAttention_9036611191413
// MI455X (gfx1250) — matrix operand order repaired and checked by reading, not compiled, not run on hardware
//
#include <hip/hip_runtime.h>
#include <hip/hip_bf16.h>

typedef __attribute__((ext_vector_type(16))) __bf16 v16bf;
typedef __attribute__((ext_vector_type(8)))  float  v8f;

#define S_LEN 2048
#define DIN   512
#define NHEAD 8
#define DK    64

__device__ __forceinline__ unsigned short f2bf(float f) {
    unsigned int u = __float_as_uint(f);
    u += 0x7FFFu + ((u >> 16) & 1u);           // round-to-nearest-even
    return (unsigned short)(u >> 16);
}
__device__ __forceinline__ unsigned int pack2(float lo, float hi) {
    return (unsigned int)f2bf(lo) | ((unsigned int)f2bf(hi) << 16);
}

union FragBF { v16bf v; uint4 q[2]; };

__device__ __forceinline__ v8f wmma_bf16(const v16bf& a, const v16bf& b, const v8f& c) {
    return __builtin_amdgcn_wmma_f32_16x16x32_bf16(false, a, false, b, (short)0, c, false, false);
}

// CDNA5 async copy: global -> LDS, 16 bytes per lane, tracked on ASYNCcnt.
// GV addressing: VDST = LDS byte offset (low 32 bits of generic pointer),
// VADDR = 64-bit global address.
__device__ __forceinline__ void async_g2l_b128(void* lds_ptr, const void* gptr) {
    unsigned lds_off = (unsigned)(unsigned long long)(uintptr_t)lds_ptr;
    unsigned long long ga = (unsigned long long)(uintptr_t)gptr;
    asm volatile("global_load_async_to_lds_b128 %0, %1, off"
                 :: "v"(lds_off), "v"(ga)
                 : "memory");
}
__device__ __forceinline__ void wait_async0() {
    asm volatile("s_wait_asynccnt 0" ::: "memory");
}

// ---------------------------------------------------------------------------
// GEMM: [8192 x 512] @ [512 x 512] + bias, 128x128 tile per block, bf16 WMMA.
// MODE 0: fp32 in  -> bf16 out, head-split   [B,H,S,Dk]
// MODE 1: fp32 in  -> bf16 out, head-split transposed [B,H,Dk,S]  (for V)
// MODE 2: bf16 in  -> f32 out = result + bias + resid (pre-LayerNorm buffer)
// ---------------------------------------------------------------------------
template<int MODE>
__global__ __launch_bounds__(256) void gemm512_kernel(
    const void* __restrict__ Ain, const float* __restrict__ W,
    const float* __restrict__ bias, const float* __restrict__ resid,
    void* __restrict__ Outv)
{
    __shared__ unsigned short As[128][40];   // A tile  [m][k], bf16, padded
    __shared__ unsigned short Bs[128][40];   // W^T tile [n][k], bf16, padded

    const int tid  = threadIdx.x;
    const int wave = tid >> 5, lane = tid & 31;
    const int l15  = lane & 15, half = lane >> 4;
    const int m0 = blockIdx.x * 128;
    const int n0 = blockIdx.y * 128;

    v8f acc[8];
    #pragma unroll
    for (int c = 0; c < 8; ++c)
        #pragma unroll
        for (int r = 0; r < 8; ++r) acc[c][r] = 0.f;

    const float*          Af = (const float*)Ain;
    const unsigned short* Ab = (const unsigned short*)Ain;

    for (int kt = 0; kt < 16; ++kt) {
        const int k0 = kt * 32;
        __syncthreads();
        if (MODE != 2) {
            #pragma unroll
            for (int i = 0; i < 4; ++i) {               // 128 x 32 fp32 -> bf16
                int slot = tid + i * 256;
                int row = slot >> 3, seg = slot & 7;
                const float4 f = *reinterpret_cast<const float4*>(
                    Af + (size_t)(m0 + row) * DIN + k0 + seg * 4);
                *reinterpret_cast<uint2*>(&As[row][seg * 4]) =
                    make_uint2(pack2(f.x, f.y), pack2(f.z, f.w));
            }
        }
        #pragma unroll
        for (int i = 0; i < 16; ++i) {                  // W^T: Bs[n][k] = W[k][n]
            int idx = i * 256 + tid;
            int kl = idx >> 7, nl = idx & 127;
            Bs[nl][kl] = f2bf(W[(size_t)(k0 + kl) * DIN + n0 + nl]);
        }
        __syncthreads();

        FragBF a;
        if (MODE == 2) {
            const unsigned short* ap =
                Ab + (size_t)(m0 + wave * 16 + l15) * DIN + k0 + half * 8;
            a.q[0] = *reinterpret_cast<const uint4*>(ap);
            a.q[1] = *reinterpret_cast<const uint4*>(ap + 16);
        } else {
            a.q[0] = *reinterpret_cast<const uint4*>(&As[wave * 16 + l15][half * 8]);
            a.q[1] = *reinterpret_cast<const uint4*>(&As[wave * 16 + l15][16 + half * 8]);
        }
        #pragma unroll
        for (int c = 0; c < 8; ++c) {
            FragBF b;
            b.q[0] = *reinterpret_cast<const uint4*>(&Bs[c * 16 + l15][half * 8]);
            b.q[1] = *reinterpret_cast<const uint4*>(&Bs[c * 16 + l15][16 + half * 8]);
            acc[c] = wmma_bf16(a.v, b.v, acc[c]);
        }
    }

    #pragma unroll
    for (int c = 0; c < 8; ++c) {
        const int n  = n0 + c * 16 + l15;
        const float bv = bias[n];
        #pragma unroll
        for (int r = 0; r < 8; ++r) {
            const int m = m0 + wave * 16 + r + 8 * half;
            float val = acc[c][r] + bv;
            if (MODE == 0) {
                int b = m >> 11, s = m & 2047, h = n >> 6, dk = n & 63;
                ((unsigned short*)Outv)[(((size_t)(b * NHEAD + h)) * S_LEN + s) * DK + dk] = f2bf(val);
            } else if (MODE == 1) {
                int b = m >> 11, s = m & 2047, h = n >> 6, dk = n & 63;
                ((unsigned short*)Outv)[(((size_t)(b * NHEAD + h)) * DK + dk) * S_LEN + s] = f2bf(val);
            } else {
                val += resid[(size_t)m * DIN + n];
                ((float*)Outv)[(size_t)m * DIN + n] = val;
            }
        }
    }
}

// ---------------------------------------------------------------------------
// Flash-attention: per (b,h) and 128-row q tile; single pass over K/V.
// scores = (Q K^T)/8 * mask; s>0 ? s : -1e4; online softmax; O = P V.
// K/V^T tiles staged with CDNA5 async global->LDS copies (ASYNCcnt).
// ---------------------------------------------------------------------------
__global__ __launch_bounds__(256) void attn_kernel(
    const unsigned short* __restrict__ Qb, const unsigned short* __restrict__ Kb,
    const unsigned short* __restrict__ VTb, const float* __restrict__ mask,
    unsigned short* __restrict__ AttnOut)
{
    __shared__ unsigned short Ks[128][72];    // K tile  [j][d]
    __shared__ unsigned short Vs[64][136];    // V^T tile [d][j]
    __shared__ unsigned short Ps[8][16][72];  // per-wave P tile (64-col half)

    const int tid  = threadIdx.x;
    const int wave = tid >> 5, lane = tid & 31;
    const int l15  = lane & 15, half = lane >> 4;
    const int qt = blockIdx.x;            // 0..15 q tiles
    const int bh = blockIdx.y;            // 0..31
    const int b = bh >> 3, h = bh & 7;

    const unsigned short* Q  = Qb  + (size_t)bh * S_LEN * DK;
    const unsigned short* K  = Kb  + (size_t)bh * S_LEN * DK;
    const unsigned short* VT = VTb + (size_t)bh * DK * S_LEN;
    const float*          Mk = mask + (size_t)b * S_LEN * S_LEN;

    const int s0w = qt * 128 + wave * 16;

    FragBF qf[2];
    #pragma unroll
    for (int kd = 0; kd < 2; ++kd) {
        const unsigned short* qp = Q + (size_t)(s0w + l15) * DK + kd * 32 + half * 8;
        qf[kd].q[0] = *reinterpret_cast<const uint4*>(qp);
        qf[kd].q[1] = *reinterpret_cast<const uint4*>(qp + 16);
    }

    float mrow[8], lrow[8];
    v8f Oacc[4];
    #pragma unroll
    for (int r = 0; r < 8; ++r) { mrow[r] = -1e30f; lrow[r] = 0.f; }
    #pragma unroll
    for (int d = 0; d < 4; ++d)
        #pragma unroll
        for (int r = 0; r < 8; ++r) Oacc[d][r] = 0.f;

    for (int jt = 0; jt < 16; ++jt) {
        const int j0 = jt * 128;
        __syncthreads();
        #pragma unroll
        for (int i = 0; i < 4; ++i) {                 // stage K 128x64 (async)
            int slot = tid + i * 256;
            int row = slot >> 3, seg = slot & 7;
            async_g2l_b128(&Ks[row][seg * 8],
                           K + (size_t)(j0 + row) * DK + seg * 8);
        }
        #pragma unroll
        for (int i = 0; i < 4; ++i) {                 // stage V^T 64x128 (async)
            int slot = tid + i * 256;
            int row = slot >> 4, seg = slot & 15;
            async_g2l_b128(&Vs[row][seg * 8],
                           VT + (size_t)row * S_LEN + j0 + seg * 8);
        }
        // prefetch next tile's mask rows into GL2 while the copies land
        if (jt < 15) {
            #pragma unroll
            for (int r = 0; r < 8; ++r)
                __builtin_prefetch(Mk + (size_t)(s0w + r + 8 * half) * S_LEN
                                      + j0 + 128 + lane * 4, 0, 0);
        }
        wait_async0();
        __syncthreads();

        v8f sc[8];
        #pragma unroll
        for (int c = 0; c < 8; ++c)
            #pragma unroll
            for (int r = 0; r < 8; ++r) sc[c][r] = 0.f;

        #pragma unroll
        for (int kd = 0; kd < 2; ++kd) {
            #pragma unroll
            for (int c = 0; c < 8; ++c) {
                FragBF kb;
                kb.q[0] = *reinterpret_cast<const uint4*>(&Ks[c * 16 + l15][kd * 32 + half * 8]);
                kb.q[1] = *reinterpret_cast<const uint4*>(&Ks[c * 16 + l15][kd * 32 + half * 8 + 16]);
                sc[c] = wmma_bf16(qf[kd].v, kb.v, sc[c]);
            }
        }

        #pragma unroll
        for (int c = 0; c < 8; ++c) {                 // scale * mask, threshold
            const int j = j0 + c * 16 + l15;
            #pragma unroll
            for (int r = 0; r < 8; ++r) {
                const int irow = s0w + r + 8 * half;
                float s = sc[c][r] * 0.125f * Mk[(size_t)irow * S_LEN + j];
                sc[c][r] = (s > 0.f) ? s : -10000.f;
            }
        }

        #pragma unroll
        for (int r = 0; r < 8; ++r) {                 // online softmax
            float tm = sc[0][r];
            #pragma unroll
            for (int c = 1; c < 8; ++c) tm = fmaxf(tm, sc[c][r]);
            #pragma unroll
            for (int off = 1; off < 16; off <<= 1) tm = fmaxf(tm, __shfl_xor(tm, off, 32));
            const float mnew  = fmaxf(mrow[r], tm);
            const float alpha = __expf(mrow[r] - mnew);
            float rs = 0.f;
            #pragma unroll
            for (int c = 0; c < 8; ++c) { float p = __expf(sc[c][r] - mnew); sc[c][r] = p; rs += p; }
            #pragma unroll
            for (int off = 1; off < 16; off <<= 1) rs += __shfl_xor(rs, off, 32);
            lrow[r] = lrow[r] * alpha + rs;
            mrow[r] = mnew;
            #pragma unroll
            for (int d = 0; d < 4; ++d) Oacc[d][r] *= alpha;
        }

        #pragma unroll
        for (int hj = 0; hj < 2; ++hj) {              // P @ V, 64-col halves
            #pragma unroll
            for (int hc = 0; hc < 4; ++hc) {
                const int c = hj * 4 + hc;
                #pragma unroll
                for (int r = 0; r < 8; ++r)
                    Ps[wave][r + 8 * half][hc * 16 + l15] = f2bf(sc[c][r]);
            }
            asm volatile("s_wait_dscnt 0" ::: "memory");  // cross-lane LDS RAW in-wave
            #pragma unroll
            for (int kh = 0; kh < 2; ++kh) {
                FragBF pa;
                pa.q[0] = *reinterpret_cast<const uint4*>(&Ps[wave][l15][kh * 32 + half * 8]);
                pa.q[1] = *reinterpret_cast<const uint4*>(&Ps[wave][l15][kh * 32 + half * 8 + 16]);
                #pragma unroll
                for (int d = 0; d < 4; ++d) {
                    FragBF vb;
                    vb.q[0] = *reinterpret_cast<const uint4*>(&Vs[d * 16 + l15][hj * 64 + kh * 32 + half * 8]);
                    vb.q[1] = *reinterpret_cast<const uint4*>(&Vs[d * 16 + l15][hj * 64 + kh * 32 + half * 8 + 16]);
                    Oacc[d] = wmma_bf16(pa.v, vb.v, Oacc[d]);
                }
            }
        }
    }

    #pragma unroll
    for (int d = 0; d < 4; ++d) {
        #pragma unroll
        for (int r = 0; r < 8; ++r) {
            const int s = s0w + r + 8 * half;
            const float v = Oacc[d][r] / lrow[r];
            AttnOut[((size_t)(b * S_LEN + s)) * DIN + h * DK + d * 16 + l15] = f2bf(v);
        }
    }
}

// ---------------------------------------------------------------------------
// LayerNorm over 512-wide rows, one block per row.
// ---------------------------------------------------------------------------
__global__ __launch_bounds__(256) void ln_kernel(
    const float* __restrict__ X, const float* __restrict__ gamma,
    const float* __restrict__ beta, float* __restrict__ out)
{
    const int row = blockIdx.x;
    const int t   = threadIdx.x;
    const float* x = X + (size_t)row * DIN;
    const float a0 = x[t], a1 = x[t + 256];
    float s  = a0 + a1;
    float s2 = a0 * a0 + a1 * a1;
    #pragma unroll
    for (int off = 16; off > 0; off >>= 1) {
        s  += __shfl_xor(s,  off, 32);
        s2 += __shfl_xor(s2, off, 32);
    }
    __shared__ float red[16];
    const int wave = t >> 5, lane = t & 31;
    if (lane == 0) { red[wave] = s; red[wave + 8] = s2; }
    __syncthreads();
    if (t == 0) {
        float ts = 0.f, ts2 = 0.f;
        for (int w = 0; w < 8; ++w) { ts += red[w]; ts2 += red[w + 8]; }
        red[0] = ts; red[8] = ts2;
    }
    __syncthreads();
    const float mu  = red[0] * (1.0f / DIN);
    const float var = red[8] * (1.0f / DIN) - mu * mu;
    const float rs  = rsqrtf(var + 1e-5f);
    out[(size_t)row * DIN + t]       = (a0 - mu) * rs * gamma[t]       + beta[t];
    out[(size_t)row * DIN + t + 256] = (a1 - mu) * rs * gamma[t + 256] + beta[t + 256];
}

extern "C" void kernel_launch(void* const* d_in, const int* in_sizes, int n_in,
                              void* d_out, int out_size, void* d_ws, size_t ws_size,
                              hipStream_t stream) {
    const float* q    = (const float*)d_in[0];
    const float* k    = (const float*)d_in[1];
    const float* v    = (const float*)d_in[2];
    const float* mask = (const float*)d_in[3];
    const float* Wq = (const float*)d_in[4];  const float* bq = (const float*)d_in[5];
    const float* Wk = (const float*)d_in[6];  const float* bk = (const float*)d_in[7];
    const float* Wv = (const float*)d_in[8];  const float* bv = (const float*)d_in[9];
    const float* Wo = (const float*)d_in[10]; const float* bo = (const float*)d_in[11];
    const float* gamma = (const float*)d_in[12];
    const float* beta  = (const float*)d_in[13];
    float* out = (float*)d_out;

    char* ws = (char*)d_ws;
    unsigned short* Qb  = (unsigned short*)(ws);                  // 8 MB bf16 [B,H,S,Dk]
    unsigned short* Kb  = (unsigned short*)(ws + (8u  << 20));    // 8 MB bf16 [B,H,S,Dk]
    unsigned short* VTb = (unsigned short*)(ws + (16u << 20));    // 8 MB bf16 [B,H,Dk,S]
    unsigned short* Ao  = (unsigned short*)(ws + (24u << 20));    // 8 MB bf16 [B*S, 512]
    float*          res = (float*)(ws + (32u << 20));             // 16 MB f32 pre-LN

    dim3 gG(64, 4), blk(256);
    gemm512_kernel<0><<<gG, blk, 0, stream>>>(q, Wq, bq, nullptr, Qb);
    gemm512_kernel<0><<<gG, blk, 0, stream>>>(k, Wk, bk, nullptr, Kb);
    gemm512_kernel<1><<<gG, blk, 0, stream>>>(v, Wv, bv, nullptr, VTb);
    attn_kernel<<<dim3(16, 32), blk, 0, stream>>>(Qb, Kb, VTb, mask, Ao);
    gemm512_kernel<2><<<gG, blk, 0, stream>>>(Ao, Wo, bo, q, res);
    ln_kernel<<<dim3(8192), blk, 0, stream>>>(res, gamma, beta, out);
}
